// Decoder_31439160607456
// MI455X (gfx1250) — matrix operand order repaired and checked by reading, not compiled, not run on hardware
//
#include <hip/hip_runtime.h>
#include <hip/hip_bf16.h>

// ---------------------------------------------------------------------------
// MI455X (gfx1250) PointNet++ decoder.
// GEMMs use v_wmma_f32_16x16x32_bf16 (wave32); each wave register-blocks
// NB o-tiles (16 x 16*NB output) and software-pipelines K with explicit
// double-buffered fragments so next-step loads overlap current-step WMMAs
// (forces distinct registers -> partial s_wait_loadcnt instead of wait-0).
// kNN-3 interpolation is fused (no B*N*S distance matrix -> stays L2 resident).
// BatchNorm is training-mode: per-channel reduce over all B*N rows.
// ---------------------------------------------------------------------------

typedef __bf16 v16bf  __attribute__((ext_vector_type(16)));
typedef __bf16 bf16x8 __attribute__((ext_vector_type(8)));
typedef float  v8f    __attribute__((ext_vector_type(8)));

__device__ __forceinline__ __bf16 f2bf(float f) {
  union { float f; unsigned u; } v; v.f = f;
  unsigned r = v.u + 0x7FFFu + ((v.u >> 16) & 1u);   // round-to-nearest-even
  unsigned short h = (unsigned short)(r >> 16);
  return __builtin_bit_cast(__bf16, h);
}

// ---------------------------------------------------------------------------
// WMMA GEMM: Y[M,Opad] = A[M,K](bf16) x W[Opad,K](bf16)^T + bias
// One 16(M) x 16*NB(O) tile per wave; K in steps of 32, double-buffered.
// Fragment layouts per CDNA5 ISA 7.12.2 (wave32):
//   A 16-bit 16x32: lanes 0-15 row=lane, elems = K{0..7,16..23};
//                   lanes 16-31 row=lane-16, elems = K{8..15,24..31}
//   B 16-bit 32x16: lane%16 = column, elems = K{0..7,16..23} / K{8..15,24..31} (same as A)
//   C/D f32 16x16 : VGPR r -> row r (+8 for lanes>=16), col = lane%16
// ---------------------------------------------------------------------------
template <int NB>
__global__ __launch_bounds__(256) void gemm_bf16_wmma(
    const __bf16* __restrict__ A, const __bf16* __restrict__ W,
    const float* __restrict__ bias, float* __restrict__ Y,
    int M, int K, int Opad)
{
  const int lane = threadIdx.x & 31;
  const int wave = blockIdx.x * (blockDim.x >> 5) + (threadIdx.x >> 5);
  const int oGrp = (Opad >> 4) / NB;       // groups of NB o-tiles
  const int mTiles = M >> 4;
  if (wave >= mTiles * oGrp) return;
  const int mt = wave / oGrp;
  const int og = wave - mt * oGrp;
  const int half = lane >> 4;              // 0: lower half-wave, 1: upper
  const int ln   = lane & 15;

  const size_t aOff = (size_t)(mt * 16 + ln) * K + half * 8;
  size_t bOff[NB];
#pragma unroll
  for (int j = 0; j < NB; ++j)
    bOff[j] = (size_t)((og * NB + j) * 16 + ln) * K + half * 8;

  v8f c[NB];
#pragma unroll
  for (int j = 0; j < NB; ++j) c[j] = v8f{};

  union frag { v16bf v; bf16x8 h[2]; };

  auto loadf = [&](int kk, frag& a, frag (&b)[NB]) {
    a.h[0] = *reinterpret_cast<const bf16x8*>(A + aOff + kk);        // K{0..7}/{8..15}
    a.h[1] = *reinterpret_cast<const bf16x8*>(A + aOff + kk + 16);   // K{16..23}/{24..31}
#pragma unroll
    for (int j = 0; j < NB; ++j) {
      b[j].h[0] = *reinterpret_cast<const bf16x8*>(W + bOff[j] + kk);
      b[j].h[1] = *reinterpret_cast<const bf16x8*>(W + bOff[j] + kk + 16);
    }
  };
  auto mmaf = [&](const frag& a, const frag (&b)[NB]) {
#pragma unroll
    for (int j = 0; j < NB; ++j)
      c[j] = __builtin_amdgcn_wmma_f32_16x16x32_bf16(
          false, a.v, false, b[j].v, (short)0, c[j], false, false);
  };

  const int steps = K >> 5;
  frag a0, a1;
  frag b0[NB], b1[NB];
  loadf(0, a0, b0);
  if (steps == 1) {                        // conv2: K = 32
    mmaf(a0, b0);
  } else {
    // steps is even (>= 2) for every layer width used here.
    int t = 0;
    for (; t + 2 < steps; t += 2) {
      loadf((t + 1) << 5, a1, b1);         // issue next-step loads first
      mmaf(a0, b0);                        //   -> waits become partial
      loadf((t + 2) << 5, a0, b0);
      mmaf(a1, b1);
    }
    loadf((t + 1) << 5, a1, b1);           // epilogue (t == steps-2)
    mmaf(a0, b0);
    mmaf(a1, b1);
  }

  const size_t rowBase = (size_t)(mt * 16 + half * 8);
#pragma unroll
  for (int j = 0; j < NB; ++j) {
    const int col = (og * NB + j) * 16 + ln;
    const float bv = bias[col];
    float* yp = Y + rowBase * Opad + col;
#pragma unroll
    for (int r = 0; r < 8; ++r) yp[(size_t)r * Opad] = c[j][r] + bv;
  }
}

// Pad+convert one weight matrix [Cout,Cin]f32 -> [Opad,Cin]bf16 and its bias.
__global__ void prep_weight(const float* __restrict__ w, __bf16* __restrict__ wb,
                            float* __restrict__ bp, const float* __restrict__ bias,
                            int Cout, int Cin, int Opad)
{
  int idx = blockIdx.x * blockDim.x + threadIdx.x;
  if (idx < Opad) bp[idx] = (idx < Cout) ? bias[idx] : 0.f;
  int total = Opad * Cin;
  if (idx >= total) return;
  int o = idx / Cin;
  wb[idx] = f2bf(o < Cout ? w[idx] : 0.f);
}

// Skip-connection transpose: f[B,C,N] -> X[(b*N+n)*Ctot + c] (bf16), cols 0..C-1
__global__ void skip_to_bf(const float* __restrict__ f, __bf16* __restrict__ X,
                           int Bn, int C, int N, int Ctot)
{
  int idx = blockIdx.x * blockDim.x + threadIdx.x;
  int total = Bn * C * N;
  if (idx >= total) return;
  int n = idx % N; int t = idx / N; int c = t % C; int b = t / C;
  X[((size_t)b * N + n) * Ctot + c] = f2bf(f[idx]);
}

// Fused kNN-3 + inverse-distance-weighted feature interpolation.
// xyzD [B,3,N] / xyzS [B,3,S] channels-first; feat strides (fB,fS,fD) generic so
// it handles both channels-first f4 and row-major [B*S,D] previous-stage output.
__global__ void knn3_interp(const float* __restrict__ xyzD, const float* __restrict__ xyzS,
                            const float* __restrict__ feat, __bf16* __restrict__ X,
                            int N, int S, int D, int Ctot, int c0,
                            long fB, long fS, long fD)
{
  extern __shared__ float sx[];                 // [3*S] source coords
  const int b = blockIdx.y;
  const float* src = xyzS + (size_t)b * 3 * S;
  for (int i = threadIdx.x; i < 3 * S; i += blockDim.x) sx[i] = src[i];
  __syncthreads();
  const int n = blockIdx.x * blockDim.x + threadIdx.x;
  if (n >= N) return;
  const size_t pb = (size_t)b * 3 * N;
  const float px = xyzD[pb + n], py = xyzD[pb + N + n], pz = xyzD[pb + 2 * N + n];

  float d0 = 3.4e38f, d1 = 3.4e38f, d2 = 3.4e38f;
  int   i0 = 0, i1 = 0, i2 = 0;
  for (int s = 0; s < S; ++s) {
    float dx = px - sx[s], dy = py - sx[S + s], dz = pz - sx[2 * S + s];
    float d = dx * dx + dy * dy + dz * dz;
    if (d < d0)      { d2 = d1; i2 = i1; d1 = d0; i1 = i0; d0 = d; i0 = s; }
    else if (d < d1) { d2 = d1; i2 = i1; d1 = d;  i1 = s; }
    else if (d < d2) { d2 = d;  i2 = s; }
  }
  float r0 = 1.f / (d0 + 1e-8f), r1 = 1.f / (d1 + 1e-8f), r2 = 1.f / (d2 + 1e-8f);
  const float inv = 1.f / (r0 + r1 + r2);
  r0 *= inv; r1 *= inv; r2 *= inv;

  const float* p0 = feat + (size_t)b * fB + (size_t)i0 * fS;
  const float* p1 = feat + (size_t)b * fB + (size_t)i1 * fS;
  const float* p2 = feat + (size_t)b * fB + (size_t)i2 * fS;
  __bf16* out = X + ((size_t)b * N + n) * Ctot + c0;
  for (int d = 0; d < D; ++d)
    out[d] = f2bf(r0 * p0[(long)d * fD] + r1 * p1[(long)d * fD] + r2 * p2[(long)d * fD]);
}

// Training-mode BN stats: per channel o, reduce sum/sumsq over M rows,
// emit fused scale/shift: y = x*sc + sh, sc = g*rsqrt(var+eps), sh = beta-mean*sc
__global__ __launch_bounds__(256) void bn_stats(
    const float* __restrict__ Y, int M, int Opad,
    const float* __restrict__ g, const float* __restrict__ beta,
    float* __restrict__ ss)
{
  __shared__ float rs[256], rq[256];
  const int o = blockIdx.x;
  float s = 0.f, q = 0.f;
  for (int m = threadIdx.x; m < M; m += 256) {
    float v = Y[(size_t)m * Opad + o];
    s += v; q += v * v;
  }
  rs[threadIdx.x] = s; rq[threadIdx.x] = q;
  __syncthreads();
  for (int st = 128; st > 0; st >>= 1) {
    if (threadIdx.x < st) {
      rs[threadIdx.x] += rs[threadIdx.x + st];
      rq[threadIdx.x] += rq[threadIdx.x + st];
    }
    __syncthreads();
  }
  if (threadIdx.x == 0) {
    float mean = rs[0] / (float)M;
    float var  = rq[0] / (float)M - mean * mean;
    float sc   = g[o] * rsqrtf(var + 1e-5f);
    ss[2 * o] = sc; ss[2 * o + 1] = beta[o] - mean * sc;
  }
}

// Apply BN + ReLU; write f32 (for next-stage interpolation) and/or bf16 (next GEMM).
__global__ void bn_apply(const float* __restrict__ Y, const float* __restrict__ ss,
                         int M, int Cout, int Opad,
                         float* __restrict__ outF, __bf16* __restrict__ outB)
{
  size_t idx = (size_t)blockIdx.x * blockDim.x + threadIdx.x;
  size_t total = (size_t)M * Cout;
  if (idx >= total) return;
  int o = (int)(idx % Cout); size_t m = idx / Cout;
  float v = fmaxf(Y[m * Opad + o] * ss[2 * o] + ss[2 * o + 1], 0.f);
  if (outF) outF[idx] = v;
  if (outB) outB[idx] = f2bf(v);
}

__global__ void log_softmax_rows(const float* __restrict__ Y, float* __restrict__ out,
                                 int M, int C, int Opad)
{
  int m = blockIdx.x * blockDim.x + threadIdx.x;
  if (m >= M) return;
  const float* r = Y + (size_t)m * Opad;
  float mx = -3.4e38f;
  for (int c = 0; c < C; ++c) mx = fmaxf(mx, r[c]);
  float s = 0.f;
  for (int c = 0; c < C; ++c) s += expf(r[c] - mx);
  float lse = mx + logf(s);
  float* o = out + (size_t)m * C;
  for (int c = 0; c < C; ++c) o[c] = r[c] - lse;
}

// ---------------------------------------------------------------------------
extern "C" void kernel_launch(void* const* d_in, const int* in_sizes, int n_in,
                              void* d_out, int out_size, void* d_ws, size_t ws_size,
                              hipStream_t stream)
{
  (void)in_sizes; (void)n_in; (void)out_size; (void)ws_size;
  const int B = 16;
  const float* xyz  = (const float*)d_in[0];
  const float* xyz1 = (const float*)d_in[1];
  const float* xyz2 = (const float*)d_in[2];
  const float* xyz3 = (const float*)d_in[3];
  const float* xyz4 = (const float*)d_in[4];
  const float* f1   = (const float*)d_in[5];
  const float* f2   = (const float*)d_in[6];
  const float* f3   = (const float*)d_in[7];
  const float* f4   = (const float*)d_in[8];

  // Param flatten order guess: top-level insertion order; nested dicts key-sorted
  // (bn1_b, bn1_g, conv1_b, conv1_w, conv2_b, conv2_w, fp1[{b,beta,g,w}x2], fp2, fp3, fp4)
  struct LP { int w, b, g, beta; int Cout, Cin, Opad; };
  const LP L[10] = {
    {18, 15, 17, 16, 256, 384, 256},   // fp1.0
    {22, 19, 21, 20, 256, 256, 256},   // fp1.1
    {26, 23, 25, 24, 256, 320, 256},   // fp2.0
    {30, 27, 29, 28, 128, 256, 128},   // fp2.1
    {34, 31, 33, 32, 128, 192, 128},   // fp3.0
    {38, 35, 37, 36, 128, 128, 128},   // fp3.1
    {42, 39, 41, 40, 128, 128, 128},   // fp4.0
    {46, 43, 45, 44,  64, 128,  64},   // fp4.1
    {12, 11, 10,  9,  32,  64,  32},   // conv1 (+bn1)
    {14, 13, -1, -1,  50,  32,  64},   // conv2 (padded to 64 cols)
  };

  char* ws = (char*)d_ws;
  size_t off = 0;
  auto carve = [&](size_t bytes) -> void* {
    void* p = ws + off; off = (off + bytes + 255) & ~(size_t)255; return p;
  };
  __bf16* WB[10]; float* BP[10];
  for (int i = 0; i < 10; ++i) {
    WB[i] = (__bf16*)carve((size_t)L[i].Opad * L[i].Cin * sizeof(__bf16));
    BP[i] = (float*)carve((size_t)L[i].Opad * sizeof(float));
  }
  float*  SS  = (float*)carve(512 * sizeof(float));
  __bf16* Abf = (__bf16*)carve((size_t)65536 * 128 * sizeof(__bf16));  // 16 MB GEMM input
  float*  Yb  = (float*)carve((size_t)65536 * 128 * sizeof(float));    // 32 MB GEMM output
  float*  FFa = (float*)carve((size_t)16384 * 128 * sizeof(float));    //  8 MB ff (odd stages)
  float*  FFb = (float*)carve((size_t)4096  * 128 * sizeof(float));    //  2 MB ff (even stages)

  // Convert/pad weights+bias to bf16 each call (deterministic, no caching).
  for (int i = 0; i < 10; ++i) {
    int total = L[i].Opad * L[i].Cin;
    prep_weight<<<(total + 255) / 256, 256, 0, stream>>>(
        (const float*)d_in[L[i].w], WB[i], BP[i], (const float*)d_in[L[i].b],
        L[i].Cout, L[i].Cin, L[i].Opad);
  }

  // Launch GEMM with the widest N-blocking the layer width allows.
  auto launch_gemm = [&](int li, int M) {
    const LP& l = L[li];
    int oT = l.Opad >> 4;
    if ((oT & 3) == 0) {
      int waves = (M >> 4) * (oT >> 2);
      gemm_bf16_wmma<4><<<(waves * 32 + 255) / 256, 256, 0, stream>>>(
          Abf, WB[li], BP[li], Yb, M, l.Cin, l.Opad);
    } else if ((oT & 1) == 0) {
      int waves = (M >> 4) * (oT >> 1);
      gemm_bf16_wmma<2><<<(waves * 32 + 255) / 256, 256, 0, stream>>>(
          Abf, WB[li], BP[li], Yb, M, l.Cin, l.Opad);
    } else {
      int waves = (M >> 4) * oT;
      gemm_bf16_wmma<1><<<(waves * 32 + 255) / 256, 256, 0, stream>>>(
          Abf, WB[li], BP[li], Yb, M, l.Cin, l.Opad);
    }
  };

  auto run_layer = [&](int li, int M, float* outF, __bf16* outB) {
    const LP& l = L[li];
    launch_gemm(li, M);
    bn_stats<<<l.Cout, 256, 0, stream>>>(Yb, M, l.Opad,
        (const float*)d_in[l.g], (const float*)d_in[l.beta], SS);
    size_t total = (size_t)M * l.Cout;
    bn_apply<<<(int)((total + 255) / 256), 256, 0, stream>>>(
        Yb, SS, M, l.Cout, l.Opad, outF, outB);
  };

  // -------- Stage 1 (fp1): N=64, S=16, skip=f3(128), interp f4(256) --------
  {
    int N = 64, S = 16, C1 = 128, Ctot = 384, D = 256;
    skip_to_bf<<<(B * C1 * N + 255) / 256, 256, 0, stream>>>(f3, Abf, B, C1, N, Ctot);
    knn3_interp<<<dim3(1, B), 64, 3 * S * sizeof(float), stream>>>(
        xyz3, xyz4, f4, Abf, N, S, D, Ctot, C1, 256L * 16L, 1L, 16L); // f4 is [B,256,16]
    run_layer(0, B * N, nullptr, Abf);   // 384->256
    run_layer(1, B * N, FFa, nullptr);   // 256->256 -> ff3 (f32)
  }
  // -------- Stage 2 (fp2): N=256, S=64, skip=f2(64), interp ff3(256) -------
  {
    int N = 256, S = 64, C1 = 64, Ctot = 320, D = 256;
    skip_to_bf<<<(B * C1 * N + 255) / 256, 256, 0, stream>>>(f2, Abf, B, C1, N, Ctot);
    knn3_interp<<<dim3(1, B), 256, 3 * S * sizeof(float), stream>>>(
        xyz2, xyz3, FFa, Abf, N, S, D, Ctot, C1, (long)S * D, (long)D, 1L);
    run_layer(2, B * N, nullptr, Abf);   // 320->256
    run_layer(3, B * N, FFb, nullptr);   // 256->128 -> ff2
  }
  // -------- Stage 3 (fp3): N=1024, S=256, skip=f1(64), interp ff2(128) -----
  {
    int N = 1024, S = 256, C1 = 64, Ctot = 192, D = 128;
    skip_to_bf<<<(B * C1 * N + 255) / 256, 256, 0, stream>>>(f1, Abf, B, C1, N, Ctot);
    knn3_interp<<<dim3(N / 256, B), 256, 3 * S * sizeof(float), stream>>>(
        xyz1, xyz2, FFb, Abf, N, S, D, Ctot, C1, (long)S * D, (long)D, 1L);
    run_layer(4, B * N, nullptr, Abf);   // 192->128
    run_layer(5, B * N, FFa, nullptr);   // 128->128 -> ff1
  }
  // -------- Stage 4 (fp4): N=4096, S=1024, no skip, interp ff1(128) --------
  {
    int N = 4096, S = 1024, Ctot = 128, D = 128;
    knn3_interp<<<dim3(N / 256, B), 256, 3 * S * sizeof(float), stream>>>(
        xyz, xyz1, FFa, Abf, N, S, D, Ctot, 0, (long)S * D, (long)D, 1L);
    int M = B * N;                        // 65536
    run_layer(6, M, nullptr, Abf);        // 128->128
    run_layer(7, M, nullptr, Abf);        // 128->64 (ff0, bf16 only)
    run_layer(8, M, nullptr, Abf);        // conv1 64->32 + bn1 + relu
    // conv2 32->50 (padded 64), no BN, then log_softmax
    {
      launch_gemm(9, M);
      log_softmax_rows<<<(M + 255) / 256, 256, 0, stream>>>(Yb, (float*)d_out, M, 50, L[9].Opad);
    }
  }
  // Second tuple output: f4 passthrough, appended after [16,4096,50] logits.
  hipMemcpyAsync((float*)d_out + (size_t)16 * 4096 * 50, f4,
                 (size_t)16 * 256 * 16 * sizeof(float),
                 hipMemcpyDeviceToDevice, stream);
}
